// GroupQueryAttention_46883863003573
// MI455X (gfx1250) — matrix operand order repaired and checked by reading, compiled, not run on hardware
//
#include <hip/hip_runtime.h>

// ---------------------------------------------------------------- constants
#define BATCH 2
#define SEQ   2048
#define DIN   1024
#define HDIM  64
#define NH    16
#define NG    4
#define NPH   (NH + 2*NG)   // 24 projection "heads": 16 Q + 4 K + 4 V

typedef __attribute__((ext_vector_type(16))) _Float16 v16h;
typedef __attribute__((ext_vector_type(8)))  _Float16 v8h;
typedef __attribute__((ext_vector_type(8)))  float    v8f;

// ------------------------------------------------------------- WMMA helper
__device__ __forceinline__ v8f wmma_f16(v16h a, v16h b, v8f c) {
  // v_wmma_f32_16x16x32_f16  (D = A*B + C), f32 accumulate
  return __builtin_amdgcn_wmma_f32_16x16x32_f16(
      /*neg_a=*/false, a, /*neg_b=*/false, b,
      /*c_mod=*/(short)0, c, /*reuse_a=*/false, /*reuse_b=*/false);
}

// Load a 16-half fragment from two 16-byte-aligned 8-half groups.
__device__ __forceinline__ v16h ldfrag(const _Float16* p0, const _Float16* p1) {
  union { v16h h; uint4 u[2]; } t;
  t.u[0] = *(const uint4*)p0;
  t.u[1] = *(const uint4*)p1;
  return t.h;
}

// Load 16 f32 (two 8-float groups) and convert to a f16 A-fragment.
__device__ __forceinline__ v16h cvtfrag(const float* p0, const float* p1) {
  float4 x0 = *(const float4*)(p0);
  float4 x1 = *(const float4*)(p0 + 4);
  float4 x2 = *(const float4*)(p1);
  float4 x3 = *(const float4*)(p1 + 4);
  v16h r;
  r[0]  = (_Float16)x0.x; r[1]  = (_Float16)x0.y;
  r[2]  = (_Float16)x0.z; r[3]  = (_Float16)x0.w;
  r[4]  = (_Float16)x1.x; r[5]  = (_Float16)x1.y;
  r[6]  = (_Float16)x1.z; r[7]  = (_Float16)x1.w;
  r[8]  = (_Float16)x2.x; r[9]  = (_Float16)x2.y;
  r[10] = (_Float16)x2.z; r[11] = (_Float16)x2.w;
  r[12] = (_Float16)x3.x; r[13] = (_Float16)x3.y;
  r[14] = (_Float16)x3.z; r[15] = (_Float16)x3.w;
  return r;
}

// ---------------------------------------------------- kernel 0: weight prep
// W (p, DIN, HDIM) f32  ->  Wt (p, HDIM, DIN) f16  (N-major for B-fragments)
__global__ __launch_bounds__(256) void prep_weights(
    const float* __restrict__ Wq, const float* __restrict__ Wk,
    const float* __restrict__ Wv, _Float16* __restrict__ Wt) {
  size_t idx = (size_t)blockIdx.x * blockDim.x + threadIdx.x;
  const size_t total = (size_t)NPH * HDIM * DIN;
  if (idx >= total) return;
  int k = (int)(idx % DIN);
  int e = (int)((idx / DIN) % HDIM);
  int p = (int)(idx / ((size_t)DIN * HDIM));
  const float* src; int lp;
  if (p < NH)            { src = Wq; lp = p; }
  else if (p < NH + NG)  { src = Wk; lp = p - NH; }
  else                   { src = Wv; lp = p - NH - NG; }
  Wt[idx] = (_Float16)src[((size_t)lp * DIN + k) * HDIM + e];
}

// ---------------------------------------------------- kernel 1: QKV project
// grid (SEQ/64, NPH, BATCH), block 128 (4 waves). Each wave: 16 rows x 64 cols.
__global__ __launch_bounds__(128) void qkv_proj(
    const float* __restrict__ query, const float* __restrict__ key,
    const float* __restrict__ value,
    const float* __restrict__ bq, const float* __restrict__ bk,
    const float* __restrict__ bv,
    const _Float16* __restrict__ Wt,
    _Float16* __restrict__ Qb,      // (B, NH, SEQ, HDIM) row-major f16
    _Float16* __restrict__ Kb,      // (B, NG, SEQ, HDIM) row-major f16
    _Float16* __restrict__ Vt) {    // (B, NG, HDIM, SEQ) transposed  f16
  const int lane = threadIdx.x & 31;
  const int w    = threadIdx.x >> 5;
  const int hi   = lane >> 4;      // 0: lanes 0-15, 1: lanes 16-31
  const int ln   = lane & 15;
  const int s0   = blockIdx.x * 64;
  const int ph   = blockIdx.y;
  const int b    = blockIdx.z;

  const float* X; const float* bias;
  if (ph < NH)           { X = query; bias = bq + ph * HDIM; }
  else if (ph < NH + NG) { X = key;   bias = bk + (ph - NH) * HDIM; }
  else                   { X = value; bias = bv + (ph - NH - NG) * HDIM; }

  const int row = s0 + w * 16 + ln;                      // A-fragment M = lane&15
  const float*    xr = X  + ((size_t)b * SEQ + row) * DIN;
  const _Float16* wt = Wt + (size_t)ph * HDIM * DIN;

  v8f acc[4] = {};                                       // 16x64 f32 tile / wave
  for (int k0 = 0; k0 < DIN; k0 += 32) {
    const int ka = k0 + hi * 8;
    v16h a = cvtfrag(xr + ka, xr + ka + 16);
#pragma unroll
    for (int c = 0; c < 4; ++c) {
      const _Float16* wp = wt + (size_t)(c * 16 + ln) * DIN + k0 + hi * 8;
      v16h bf = ldfrag(wp, wp + 16);
      acc[c] = wmma_f16(a, bf, acc[c]);
    }
  }

  if (ph < NH + NG) {   // Q and K: row-major (seq, dim)
    _Float16* dst = (ph < NH)
        ? Qb + ((size_t)b * NH + ph)        * SEQ * HDIM
        : Kb + ((size_t)b * NG + (ph - NH)) * SEQ * HDIM;
#pragma unroll
    for (int c = 0; c < 4; ++c) {
      const float bs = bias[c * 16 + ln];
#pragma unroll
      for (int r = 0; r < 8; ++r) {
        const int m = w * 16 + hi * 8 + r;               // C layout: M = r + 8*hi
        dst[(size_t)(s0 + m) * HDIM + c * 16 + ln] = (_Float16)(acc[c][r] + bs);
      }
    }
  } else {              // V: transposed (dim, seq) -> contiguous 8-half stores
    const int g = ph - NH - NG;
#pragma unroll
    for (int c = 0; c < 4; ++c) {
      const float bs = bias[c * 16 + ln];
      const int n = c * 16 + ln;
      v8h pk;
#pragma unroll
      for (int r = 0; r < 8; ++r) pk[r] = (_Float16)(acc[c][r] + bs);
      *(v8h*)(Vt + (((size_t)b * NG + g) * HDIM + n) * SEQ + s0 + w * 16 + hi * 8) = pk;
    }
  }
}

// ---------------------------------------------------- kernel 2: flash attn
// grid (SEQ/64, NH, BATCH), block 128 (4 waves). Each wave: 16 query rows.
__global__ __launch_bounds__(128) void gqa_attn(
    const _Float16* __restrict__ Qb, const _Float16* __restrict__ Kb,
    const _Float16* __restrict__ Vt, float* __restrict__ out) {
  __shared__ __align__(16) _Float16 plds[4][16][72];   // per-wave P tile, padded

  const int lane = threadIdx.x & 31;
  const int w    = threadIdx.x >> 5;
  const int hi   = lane >> 4;
  const int ln   = lane & 15;
  const int q0   = blockIdx.x * 64;
  const int h    = blockIdx.y;
  const int b    = blockIdx.z;
  const int g    = h & (NG - 1);                       // h % G

  // Resident Q A-fragments (dims 0..31 and 32..63)
  const int qrow = q0 + w * 16 + ln;
  const _Float16* qp = Qb + (((size_t)b * NH + h) * SEQ + qrow) * HDIM;
  const int d0 = hi * 8;
  v16h a0 = ldfrag(qp + d0,      qp + d0 + 16);
  v16h a1 = ldfrag(qp + d0 + 32, qp + d0 + 48);

  const _Float16* kbase = Kb + ((size_t)b * NG + g) * SEQ * HDIM;
  const _Float16* vbase = Vt + ((size_t)b * NG + g) * HDIM * SEQ;

  v8f acc[4] = {};
  float mrow[8], lrow[8];
#pragma unroll
  for (int r = 0; r < 8; ++r) { mrow[r] = -1e30f; lrow[r] = 0.0f; }

  const int nkb = q0 / 64 + 1;                         // causal: key blocks <= q block
  for (int kb64 = 0; kb64 < nkb; ++kb64) {
    const int kb = kb64 * 64;

    // ---- scores: 4 fragments of 16 keys each
    v8f s[4];
#pragma unroll
    for (int c = 0; c < 4; ++c) {
      const _Float16* kp = kbase + (size_t)(kb + c * 16 + ln) * HDIM + hi * 8;
      v16h b0 = ldfrag(kp,      kp + 16);              // dims  0..31 slice
      v16h b1 = ldfrag(kp + 32, kp + 48);              // dims 32..63 slice
      v8f z = {};
      z = wmma_f16(a0, b0, z);
      z = wmma_f16(a1, b1, z);
      const int kidx = kb + c * 16 + ln;
#pragma unroll
      for (int r = 0; r < 8; ++r) {
        const int qidx = q0 + w * 16 + hi * 8 + r;
        const float v  = z[r] * 0.125f;                // 1/sqrt(64)
        s[c][r] = (kidx <= qidx) ? v : -1e9f;          // causal mask
      }
    }

    // ---- online softmax: row max over 64 keys (16-lane xor reduction)
    float scl[8];
#pragma unroll
    for (int r = 0; r < 8; ++r) {
      float v = fmaxf(fmaxf(s[0][r], s[1][r]), fmaxf(s[2][r], s[3][r]));
      v = fmaxf(v, __shfl_xor(v, 1, 32));
      v = fmaxf(v, __shfl_xor(v, 2, 32));
      v = fmaxf(v, __shfl_xor(v, 4, 32));
      v = fmaxf(v, __shfl_xor(v, 8, 32));
      const float mn = fmaxf(mrow[r], v);
      scl[r]  = __expf(mrow[r] - mn);
      mrow[r] = mn;
    }
    float ls[8];
#pragma unroll
    for (int r = 0; r < 8; ++r) ls[r] = 0.0f;
#pragma unroll
    for (int c = 0; c < 4; ++c)
#pragma unroll
      for (int r = 0; r < 8; ++r) {
        const float p = __expf(s[c][r] - mrow[r]);
        s[c][r] = p;
        ls[r] += p;
      }
#pragma unroll
    for (int r = 0; r < 8; ++r) {
      float v = ls[r];
      v += __shfl_xor(v, 1, 32);
      v += __shfl_xor(v, 2, 32);
      v += __shfl_xor(v, 4, 32);
      v += __shfl_xor(v, 8, 32);
      lrow[r] = lrow[r] * scl[r] + v;
    }
#pragma unroll
    for (int c = 0; c < 4; ++c)
#pragma unroll
      for (int r = 0; r < 8; ++r) acc[c][r] *= scl[r];

    // ---- P: C-layout -> LDS -> A-layout (per-wave private region)
#pragma unroll
    for (int c = 0; c < 4; ++c)
#pragma unroll
      for (int r = 0; r < 8; ++r)
        plds[w][hi * 8 + r][c * 16 + ln] = (_Float16)s[c][r];

    // ---- PV: 2 key chunks x 4 dim chunks
#pragma unroll
    for (int t = 0; t < 2; ++t) {
      const _Float16* pp = &plds[w][ln][t * 32 + hi * 8];
      v16h pa = ldfrag(pp, pp + 16);
#pragma unroll
      for (int c = 0; c < 4; ++c) {
        const _Float16* vp =
            vbase + (size_t)(c * 16 + ln) * SEQ + kb + t * 32 + hi * 8;
        v16h vb = ldfrag(vp, vp + 16);
        acc[c] = wmma_f16(pa, vb, acc[c]);
      }
    }
  }

  // ---- epilogue: out[b][s][h*64 + e] = acc / l
#pragma unroll
  for (int c = 0; c < 4; ++c) {
#pragma unroll
    for (int r = 0; r < 8; ++r) {
      const int srow = q0 + w * 16 + hi * 8 + r;
      out[((size_t)b * SEQ + srow) * (NH * HDIM) + h * HDIM + c * 16 + ln] =
          acc[c][r] / lrow[r];
    }
  }
}

// ------------------------------------------------------------------ launch
extern "C" void kernel_launch(void* const* d_in, const int* in_sizes, int n_in,
                              void* d_out, int out_size, void* d_ws, size_t ws_size,
                              hipStream_t stream) {
  const float* query = (const float*)d_in[0];
  const float* key   = (const float*)d_in[1];
  const float* value = (const float*)d_in[2];
  // d_in[3] = tril mask (int32) — causality computed from indices instead.
  const float* Wq = (const float*)d_in[4];
  const float* bq = (const float*)d_in[5];
  const float* Wk = (const float*)d_in[6];
  const float* bk = (const float*)d_in[7];
  const float* Wv = (const float*)d_in[8];
  const float* bv = (const float*)d_in[9];
  float* out = (float*)d_out;

  // Workspace carve-up (f16 halves):
  _Float16* ws = (_Float16*)d_ws;
  _Float16* Wt = ws;                                           // 24*64*1024
  _Float16* Qb = Wt + (size_t)NPH * HDIM * DIN;                // 2*16*2048*64
  _Float16* Kb = Qb + (size_t)BATCH * NH * SEQ * HDIM;         // 2*4*2048*64
  _Float16* Vt = Kb + (size_t)BATCH * NG * SEQ * HDIM;         // 2*4*64*2048

  const int wtotal = NPH * HDIM * DIN;
  prep_weights<<<(wtotal + 255) / 256, 256, 0, stream>>>(Wq, Wk, Wv, Wt);

  dim3 g1(SEQ / 64, NPH, BATCH);
  qkv_proj<<<g1, 128, 0, stream>>>(query, key, value, bq, bk, bv, Wt, Qb, Kb, Vt);

  dim3 g2(SEQ / 64, NH, BATCH);
  gqa_attn<<<g2, 128, 0, stream>>>(Qb, Kb, Vt, out);
}
